// NoisyGNN_12068858102169
// MI455X (gfx1250) — matrix operand order repaired and checked by reading, not compiled, not run on hardware
//
#include <hip/hip_runtime.h>
#include <hip/hip_bf16.h>

typedef __attribute__((ext_vector_type(16))) _Float16 v16h;
typedef __attribute__((ext_vector_type(8)))  _Float16 v8h;
typedef __attribute__((ext_vector_type(4)))  _Float16 v4h;
typedef __attribute__((ext_vector_type(8)))  float    v8f;
typedef __attribute__((ext_vector_type(4)))  float    v4f;
typedef __attribute__((ext_vector_type(4)))  unsigned int v4u;
typedef __attribute__((ext_vector_type(8)))  int      v8i;
typedef __attribute__((ext_vector_type(4)))  int      v4i;

#define N_NODES 50000
#define N_EDGES 800000
#define D_FEAT  128
#define D_OUT   40
#define ROW_TILES 3125  // 50000 / 16 exactly

#if __has_builtin(__builtin_amdgcn_tensor_load_to_lds) && \
    __has_builtin(__builtin_amdgcn_s_wait_tensorcnt)
#define HAVE_TDM 1
#else
#define HAVE_TDM 0
#endif

// ---------------------------------------------------------------------------
// Utility: zero a float buffer
// ---------------------------------------------------------------------------
__global__ void zero_f32_kernel(float* __restrict__ p, int n) {
    int gid = blockIdx.x * blockDim.x + threadIdx.x;
    if (gid < n) p[gid] = 0.0f;
}

// ---------------------------------------------------------------------------
// Degree: one thread per edge, atomic scatter of 1.0 onto dst
// ---------------------------------------------------------------------------
__global__ void degree_kernel(const long long* __restrict__ dst,
                              float* __restrict__ deg, int n_edges) {
    int e = blockIdx.x * blockDim.x + threadIdx.x;
    if (e < n_edges) {
        atomicAdd(&deg[(int)dst[e]], 1.0f);
    }
}

__global__ void inv_deg_kernel(const float* __restrict__ deg,
                               float* __restrict__ inv_deg, int n) {
    int i = blockIdx.x * blockDim.x + threadIdx.x;
    if (i < n) {
        float d = deg[i];
        inv_deg[i] = (d > 0.0f) ? (1.0f / d) : 0.0f;
    }
}

// ---------------------------------------------------------------------------
// Edge-parallel scatter-add of features: agg[dst] += feat[src]
// 32 threads per edge, 4 contiguous floats each; global_atomic_add_f32
// scatter (the 25.6MB feature matrix lives in the 192MB L2).
// ---------------------------------------------------------------------------
__global__ void scatter_feat_kernel(const float* __restrict__ feat,
                                    const long long* __restrict__ src,
                                    const long long* __restrict__ dst,
                                    float* __restrict__ agg, int n_edges) {
    int gid = blockIdx.x * blockDim.x + threadIdx.x;
    int e = gid >> 5;
    if (e >= n_edges) return;
    int c = (gid & 31) * 4;
    const float* fr = feat + (size_t)src[e] * D_FEAT + c;
    float*       to = agg  + (size_t)dst[e] * D_FEAT + c;
    v4f v = *(const v4f*)fr;
    atomicAdd(&to[0], v.x);
    atomicAdd(&to[1], v.y);
    atomicAdd(&to[2], v.z);
    atomicAdd(&to[3], v.w);
}

// ---------------------------------------------------------------------------
// Scale by inv_deg (mean aggregation) and convert f32 -> f16 for WMMA A-matrix
// ---------------------------------------------------------------------------
__global__ void scale_cvt_kernel(const float* __restrict__ agg,
                                 const float* __restrict__ inv_deg,
                                 _Float16* __restrict__ a16, int n_nodes) {
    int gid = blockIdx.x * blockDim.x + threadIdx.x;
    int node = gid >> 5;
    if (node >= n_nodes) return;
    int c = (gid & 31) * 4;
    float s = inv_deg[node];
    v4f v = *(const v4f*)(agg + (size_t)node * D_FEAT + c);
    v4h h;
    h.x = (_Float16)(v.x * s);
    h.y = (_Float16)(v.y * s);
    h.z = (_Float16)(v.z * s);
    h.w = (_Float16)(v.w * s);
    *(v4h*)(a16 + (size_t)node * D_FEAT + c) = h;
}

// ---------------------------------------------------------------------------
// One-time weight prep: f32 [cols_real,128] -> padded f16 image [NCT*16,128]
// ---------------------------------------------------------------------------
__global__ void cvt_w_kernel(const float* __restrict__ W,
                             _Float16* __restrict__ W16,
                             int cols_real, int total) {
    int idx = blockIdx.x * blockDim.x + threadIdx.x;
    if (idx >= total) return;
    int col = idx >> 7;
    int k   = idx & 127;
    W16[idx] = (col < cols_real) ? (_Float16)W[col * 128 + k] : (_Float16)0.0f;
}

// ---------------------------------------------------------------------------
// WMMA GEMM:  out[m, o] = sum_k A16[m, k] * W16[o, k]   (A @ W^T, K = 128)
//   - W16 is the pre-padded f16 weight image [NCT*16, 128]; staged into LDS
//     by the Tensor Data Mover (wave 0 issues tensor_load_to_lds with a 1-D
//     D# descriptor, waits on TENSORcnt, then the block barriers).
//   - Each wave owns one 16-row tile, sweeps NCT column tiles, 4 K-steps of
//     v_wmma_f32_16x16x32_f16 each, straight-line epilogue (templated flags).
// ---------------------------------------------------------------------------
template <int NCT, int NCOLS, bool RELU, bool EMIT16>
__global__ __launch_bounds__(256)
void gemm_wmma_kernel(const _Float16* __restrict__ a16,
                      const _Float16* __restrict__ w16,
                      float* __restrict__ out32, int ld_out,
                      _Float16* __restrict__ out16) {
    __shared__ alignas(32) _Float16 Wlds[NCT * 16 * 128];
    constexpr unsigned WBYTES = (unsigned)NCT * 16u * 128u * 2u;

#if HAVE_TDM
    if (threadIdx.x < 32) {
        // ---- 1-D TDM descriptor: copy WBYTES bytes (8B elements) G->LDS ----
        const unsigned n8 = WBYTES / 8u;   // elements of data_size=8B
        unsigned long long gaddr = (unsigned long long)w16;
        unsigned lds_addr = (unsigned)(unsigned long long)(void*)Wlds;
        v4u g0;
        g0[0] = 1u;                                   // count=1, user mode
        g0[1] = lds_addr;                             // lds_addr [63:32]
        g0[2] = (unsigned)(gaddr & 0xFFFFFFFFu);      // global_addr lo
        g0[3] = (unsigned)((gaddr >> 32) & 0x1FFFFFFu) | (2u << 30); // hi | type=2
        v8i g1 = {};
        g1[0] = (int)(3u << 16);                      // data_size = 8B
        g1[1] = (int)((n8 & 0xFFFFu) << 16);          // tensor_dim0[15:0] @63:48
        g1[2] = (int)(((n8 >> 16) & 0xFFFFu) | (1u << 16)); // dim0 hi | tensor_dim1=1
        g1[3] = (int)((n8 & 0xFFFFu) << 16);          // tile_dim0 @127:112
        g1[5] = (int)n8;                              // tensor_dim0_stride lo
        v4i g2 = {};
        v4i g3 = {};
#if defined(__clang_major__) && (__clang_major__ >= 23)
        v8i g4 = {};
        __builtin_amdgcn_tensor_load_to_lds(g0, g1, g2, g3, g4, 0);
#else
        __builtin_amdgcn_tensor_load_to_lds(g0, g1, g2, g3, 0);
#endif
        __builtin_amdgcn_s_wait_tensorcnt(0);
    }
#else
    for (unsigned idx = threadIdx.x; idx < WBYTES / 16u; idx += 256u)
        ((v8h*)Wlds)[idx] = ((const v8h*)w16)[idx];
#endif
    __syncthreads();

    const int tid  = threadIdx.x;
    const int wave = tid >> 5;
    const int lane = tid & 31;
    const int rt   = blockIdx.x * 8 + wave;   // 16-row tile index
    if (rt >= ROW_TILES) return;

    const int m0 = lane & 15;   // A row within tile / B,C column within tile
    const int hi = lane >> 4;   // lane-half selector

    // A fragments (16x32 f16): lane m holds K[kb..kb+15],
    // kb = hi*16 within each 32-wide K step.
    v16h afr[4];
    const size_t arow = (size_t)(rt * 16 + m0) * D_FEAT;
#pragma unroll
    for (int kt = 0; kt < 4; ++kt) {
        const int kb = kt * 32 + hi * 16;
        v8h lo = *(const v8h*)(a16 + arow + kb);
        v8h hh = *(const v8h*)(a16 + arow + kb + 8);
        v16h a;
#pragma unroll
        for (int i = 0; i < 8; ++i) { a[i] = lo[i]; a[i + 8] = hh[i]; }
        afr[kt] = a;
    }

#pragma unroll
    for (int ct = 0; ct < NCT; ++ct) {
        v8f acc = {};
#pragma unroll
        for (int kt = 0; kt < 4; ++kt) {
            // B fragment 32x16: lane holds column n=m0, K range hi*16..hi*16+15
            // of this K-step -> one 32B contiguous ds_load per lane.
            const _Float16* bp = &Wlds[(ct * 16 + m0) * 128 + kt * 32 + hi * 16];
            v16h b = *(const v16h*)bp;
            acc = __builtin_amdgcn_wmma_f32_16x16x32_f16(
                false, afr[kt], false, b, (short)0, acc, false, false);
        }
        // C/D layout: VGPR j, lane l -> M = hi*8 + j, N = m0.
        const int n = ct * 16 + m0;
        if ((NCOLS % 16 == 0) || (n < NCOLS)) {
            const int mbase = rt * 16 + hi * 8;
            float* op = out32 + (size_t)mbase * ld_out + n;
#pragma unroll
            for (int j = 0; j < 8; ++j) {
                float v = acc[j];
                if (RELU) v = v > 0.0f ? v : 0.0f;
                op[(size_t)j * ld_out] = v;
                if (EMIT16) out16[(size_t)(mbase + j) * D_FEAT + n] = (_Float16)v;
            }
        }
    }
}

// ---------------------------------------------------------------------------
// Row-wise log-softmax over 40 logits (+bias), logits stored with ld=48
// ---------------------------------------------------------------------------
__global__ void log_softmax_kernel(const float* __restrict__ logits,
                                   const float* __restrict__ bias,
                                   float* __restrict__ out, int n_rows) {
    int r = blockIdx.x * blockDim.x + threadIdx.x;
    if (r >= n_rows) return;
    const float* row = logits + (size_t)r * 48;
    float v[D_OUT];
    float mx = -3.4e38f;
#pragma unroll
    for (int j = 0; j < D_OUT; ++j) {
        v[j] = row[j] + bias[j];
        mx = v[j] > mx ? v[j] : mx;
    }
    float s = 0.0f;
#pragma unroll
    for (int j = 0; j < D_OUT; ++j) s += __expf(v[j] - mx);
    float lse = mx + __logf(s);
#pragma unroll
    for (int j = 0; j < D_OUT; ++j) out[(size_t)r * D_OUT + j] = v[j] - lse;
}

// ---------------------------------------------------------------------------
// Host-side launch sequence (graph-capture safe: only kernel launches)
// ---------------------------------------------------------------------------
extern "C" void kernel_launch(void* const* d_in, const int* in_sizes, int n_in,
                              void* d_out, int out_size, void* d_ws, size_t ws_size,
                              hipStream_t stream) {
    const float*     x    = (const float*)d_in[0];
    const long long* edge = (const long long*)d_in[1];   // int64 [2, E]
    const float*     W1   = (const float*)d_in[2];
    const float*     W2   = (const float*)d_in[3];
    const float*     W3   = (const float*)d_in[4];
    const float*     outW = (const float*)d_in[5];
    const float*     outB = (const float*)d_in[6];
    (void)in_sizes; (void)n_in; (void)out_size; (void)ws_size;

    const long long* src = edge;
    const long long* dst = edge + N_EDGES;

    // Workspace carve-up (all offsets 16B-aligned)
    float* deg     = (float*)d_ws;                       // 50048
    float* inv_deg = deg + 50048;                        // 50048
    float* agg     = inv_deg + 50048;                    // N*128
    _Float16* a16  = (_Float16*)(agg + (size_t)N_NODES * D_FEAT);  // N*128 f16
    float* hbuf    = (float*)(a16 + (size_t)N_NODES * D_FEAT);     // N*128
    float* logits  = hbuf + (size_t)N_NODES * D_FEAT;              // N*48
    _Float16* w1h  = (_Float16*)(logits + (size_t)N_NODES * 48);   // 128*128
    _Float16* w2h  = w1h + 128 * 128;
    _Float16* w3h  = w2h + 128 * 128;
    _Float16* woh  = w3h + 128 * 128;                              // 48*128

    float* d_logp   = (float*)d_out;                     // [N, 40]
    float* d_hidden = d_logp + (size_t)N_NODES * D_OUT;  // [N, 128]

    const int ZB = 256;
    const int n_feat_elems = N_NODES * D_FEAT;
    const int scat_threads = N_EDGES * 32;
    const int node_threads = N_NODES * 32;
    const int gemm_blocks  = (ROW_TILES + 7) / 8;

    // Degrees
    zero_f32_kernel<<<(N_NODES + ZB - 1) / ZB, ZB, 0, stream>>>(deg, N_NODES);
    degree_kernel<<<(N_EDGES + ZB - 1) / ZB, ZB, 0, stream>>>(dst, deg, N_EDGES);
    inv_deg_kernel<<<(N_NODES + ZB - 1) / ZB, ZB, 0, stream>>>(deg, inv_deg, N_NODES);

    // One-time f16 weight images (padded column tiles) for TDM staging
    cvt_w_kernel<<<(16384 + ZB - 1) / ZB, ZB, 0, stream>>>(W1, w1h, 128, 16384);
    cvt_w_kernel<<<(16384 + ZB - 1) / ZB, ZB, 0, stream>>>(W2, w2h, 128, 16384);
    cvt_w_kernel<<<(16384 + ZB - 1) / ZB, ZB, 0, stream>>>(W3, w3h, 128, 16384);
    cvt_w_kernel<<<(6144  + ZB - 1) / ZB, ZB, 0, stream>>>(outW, woh, D_OUT, 6144);

    // ---- Layer 1: h = relu(mean_agg(x) @ W1^T) ----
    zero_f32_kernel<<<(n_feat_elems + ZB - 1) / ZB, ZB, 0, stream>>>(agg, n_feat_elems);
    scatter_feat_kernel<<<(scat_threads + ZB - 1) / ZB, ZB, 0, stream>>>(x, src, dst, agg, N_EDGES);
    scale_cvt_kernel<<<(node_threads + ZB - 1) / ZB, ZB, 0, stream>>>(agg, inv_deg, a16, N_NODES);
    gemm_wmma_kernel<8, 128, true, false><<<gemm_blocks, 256, 0, stream>>>(
        a16, w1h, hbuf, 128, nullptr);

    // ---- Layer 2: h = relu(mean_agg(h) @ W2^T) ----
    zero_f32_kernel<<<(n_feat_elems + ZB - 1) / ZB, ZB, 0, stream>>>(agg, n_feat_elems);
    scatter_feat_kernel<<<(scat_threads + ZB - 1) / ZB, ZB, 0, stream>>>(hbuf, src, dst, agg, N_EDGES);
    scale_cvt_kernel<<<(node_threads + ZB - 1) / ZB, ZB, 0, stream>>>(agg, inv_deg, a16, N_NODES);
    gemm_wmma_kernel<8, 128, true, false><<<gemm_blocks, 256, 0, stream>>>(
        a16, w2h, hbuf, 128, nullptr);

    // ---- Layer 3: hidden = mean_agg(h) @ W3^T (no relu); emit f16 copy too.
    // In-place f16 re-emit into a16 is safe: each wave reads its own 16 rows
    // fully into registers before its epilogue writes those same rows. ----
    zero_f32_kernel<<<(n_feat_elems + ZB - 1) / ZB, ZB, 0, stream>>>(agg, n_feat_elems);
    scatter_feat_kernel<<<(scat_threads + ZB - 1) / ZB, ZB, 0, stream>>>(hbuf, src, dst, agg, N_EDGES);
    scale_cvt_kernel<<<(node_threads + ZB - 1) / ZB, ZB, 0, stream>>>(agg, inv_deg, a16, N_NODES);
    gemm_wmma_kernel<8, 128, false, true><<<gemm_blocks, 256, 0, stream>>>(
        a16, w3h, d_hidden, 128, a16);

    // ---- Output projection: logits = hidden @ outW^T (40 cols, padded to 48)
    gemm_wmma_kernel<3, 40, false, false><<<gemm_blocks, 256, 0, stream>>>(
        a16, woh, logits, 48, nullptr);

    // ---- log_softmax (+ bias) ----
    log_softmax_kernel<<<(N_NODES + ZB - 1) / ZB, ZB, 0, stream>>>(logits, outB, d_logp, N_NODES);
}
